// LogPolarConvolution_84473416778309
// MI455X (gfx1250) — matrix operand order repaired and checked by reading, not compiled, not run on hardware
//
#include <hip/hip_runtime.h>
#include <math.h>

typedef __attribute__((ext_vector_type(16))) __bf16 v16bf;
typedef __attribute__((ext_vector_type(8)))  __bf16 v8bf;
typedef __attribute__((ext_vector_type(8)))  float  v8f;

#define BB   16
#define CC   64
#define HH   256
#define WW   256
#define HWP  258           // padded H/W
#define HW   (HH*WW)       // 65536
#define KTOT 576           // 9 taps * 64 ci
#define PI_F 3.14159265358979323846f
#define EPSF 0.001f

__device__ __forceinline__ unsigned short f2bf(float f) {
  union { float f; unsigned u; } v; v.f = f;
  unsigned r = v.u + 0x7FFFu + ((v.u >> 16) & 1u);   // round-to-nearest-even
  return (unsigned short)(r >> 16);
}
__device__ __forceinline__ float bf2f(unsigned short u) {
  union { unsigned v; float f; } x; x.v = ((unsigned)u) << 16; return x.f;
}

// ---------------------------------------------------------------------------
// Weight transform: Wt[n][k] bf16, k = (kh*3+kw)*64 + ci   (OIHW f32 input)
// ---------------------------------------------------------------------------
__global__ void k_prep_w(const float* __restrict__ w, unsigned short* __restrict__ wt) {
  int t = blockIdx.x * blockDim.x + threadIdx.x;         // over 64*576
  if (t >= CC * KTOT) return;
  int n  = t / KTOT;
  int k  = t - n * KTOT;
  int tap = k >> 6;            // kh*3+kw
  int ci  = k & 63;
  int kh  = tap / 3, kw = tap - kh * 3;
  wt[t] = f2bf(w[((n * CC + ci) * 3 + kh) * 3 + kw]);
}

// ---------------------------------------------------------------------------
// cart -> log-polar resample, write zero-padded NHWC bf16: lp[b][hp][wp][c]
// ---------------------------------------------------------------------------
__global__ void k_cart2lp(const float* __restrict__ x, unsigned short* __restrict__ lp) {
  int t = blockIdx.x * blockDim.x + threadIdx.x;         // over B*258*258
  if (t >= BB * HWP * HWP) return;
  int b  = t / (HWP * HWP);
  int rm = t - b * (HWP * HWP);
  int hp = rm / HWP;
  int wp = rm - hp * HWP;
  int obase = t * CC;

  if (hp == 0 || hp > HH || wp == 0 || wp > WW) {        // zero border
    uint4 z = make_uint4(0u, 0u, 0u, 0u);
    #pragma unroll
    for (int cb = 0; cb < CC; cb += 8)
      *(uint4*)(lp + obase + cb) = z;
    return;
  }
  int h = hp - 1, w = wp - 1;
  const float cx = (WW - 1) * 0.5f, cy = (HH - 1) * 0.5f;
  const float rmax = sqrtf(128.5f * 128.5f * 2.0f);
  const float logrm = logf(rmax + EPSF);
  float logr  = (float)w * (logrm / (float)WW);
  float theta = (float)h * (2.0f * PI_F / (float)HH);
  float r  = expf(logr);
  float gx = r * cosf(theta) + cx;
  float gy = r * sinf(theta) + cy;

  float left = floorf(gx), top = floorf(gy);
  int li = (int)left, ti = (int)top, ri = li + 1, bi = ti + 1;
  float dl = gx - left, dr = left + 1.0f - gx;
  float dt = gy - top,  db = top + 1.0f - gy;
  float wtl = db * dr, wtr = db * dl, wbl = dt * dr, wbr = dt * dl;

  bool vtl = (li >= 0) & (li < WW) & (ti >= 0) & (ti < HH);
  bool vtr = (ri >= 0) & (ri < WW) & (ti >= 0) & (ti < HH);
  bool vbl = (li >= 0) & (li < WW) & (bi >= 0) & (bi < HH);
  bool vbr = (ri >= 0) & (ri < WW) & (bi >= 0) & (bi < HH);
  int otl = vtl ? (ti * WW + li) : 0;
  int otr = vtr ? (ti * WW + ri) : 0;
  int obl = vbl ? (bi * WW + li) : 0;
  int obr = vbr ? (bi * WW + ri) : 0;
  float ftl = vtl ? wtl : 0.0f, ftr = vtr ? wtr : 0.0f;
  float fbl = vbl ? wbl : 0.0f, fbr = vbr ? wbr : 0.0f;

  int pb = b * CC * HW;
  #pragma unroll
  for (int cb = 0; cb < CC; cb += 8) {
    unsigned pk[4];
    #pragma unroll
    for (int q = 0; q < 8; q += 2) {
      int p0 = pb + (cb + q) * HW, p1 = p0 + HW;
      float v0 = ftl * x[p0 + otl] + ftr * x[p0 + otr] + fbl * x[p0 + obl] + fbr * x[p0 + obr];
      float v1 = ftl * x[p1 + otl] + ftr * x[p1 + otr] + fbl * x[p1 + obl] + fbr * x[p1 + obr];
      pk[q >> 1] = (unsigned)f2bf(v0) | ((unsigned)f2bf(v1) << 16);
    }
    *(uint4*)(lp + obase + cb) = make_uint4(pk[0], pk[1], pk[2], pk[3]);
  }
}

// ---------------------------------------------------------------------------
// 3x3 conv as implicit GEMM with v_wmma_f32_16x16x32_bf16.
// One wave: 16 spatial (consecutive w) x 64 out-channels; K = 576 in 18 steps.
// lp: padded NHWC bf16, wt: [n][k] bf16, y out: NCHW bf16.
// ---------------------------------------------------------------------------
__global__ void k_conv_wmma(const unsigned short* __restrict__ lp,
                            const unsigned short* __restrict__ wt,
                            const float* __restrict__ bias,
                            unsigned short* __restrict__ y) {
  const int lane = threadIdx.x & 31;
  const int wv   = blockIdx.x * (blockDim.x >> 5) + (threadIdx.x >> 5);
  const int b    = wv >> 12;            // 4096 tiles per image
  const int tile = wv & 4095;
  const int m0   = tile << 4;
  const int h    = m0 >> 8;
  const int w0   = m0 & 255;
  const int r    = lane & 15;
  const int hi   = lane >> 4;

  union AV { v16bf v; v8bf h[2]; };

  v8f acc[4];
  #pragma unroll
  for (int j = 0; j < 4; ++j) {
    float bv = bias[r + j * 16];
    acc[j] = (v8f){bv, bv, bv, bv, bv, bv, bv, bv};
  }

  for (int kk = 0; kk < 18; ++kk) {
    const int tap = kk >> 1;                 // 0..8
    const int p   = kk & 1;                  // channel half (32 ch each)
    const int dh  = tap / 3 - 1;
    const int dw  = tap - (tap / 3) * 3 - 1;
    const int hp  = h + 1 + dh;
    const int wp  = w0 + r + 1 + dw;
    const int aidx = ((b * HWP + hp) * HWP + wp) * CC + p * 32 + hi * 8;
    AV a;
    a.h[0] = *(const v8bf*)(lp + aidx);          // K halves 0..7  of this lane
    a.h[1] = *(const v8bf*)(lp + aidx + 16);     // K halves 8..15 of this lane

    const int kbase = kk * 32 + hi * 8;
    #pragma unroll
    for (int j = 0; j < 4; ++j) {
      const int n = r + j * 16;
      AV bm;
      bm.h[0] = *(const v8bf*)(wt + n * KTOT + kbase);
      bm.h[1] = *(const v8bf*)(wt + n * KTOT + kbase + 16);
      acc[j] = __builtin_amdgcn_wmma_f32_16x16x32_bf16(
          false, a.v, false, bm.v, (short)0, acc[j], false, false);
    }
  }

  // D layout: VGPR i -> M = i + 8*hi, lane -> N = r + 16*j. 8 consecutive m.
  #pragma unroll
  for (int j = 0; j < 4; ++j) {
    const int n = r + j * 16;
    unsigned pk[4];
    #pragma unroll
    for (int q = 0; q < 8; q += 2)
      pk[q >> 1] = (unsigned)f2bf(acc[j][q]) | ((unsigned)f2bf(acc[j][q + 1]) << 16);
    const int oidx = (b * CC + n) * HW + m0 + hi * 8;
    *(uint4*)(y + oidx) = make_uint4(pk[0], pk[1], pk[2], pk[3]);
  }
}

// ---------------------------------------------------------------------------
// log-polar -> cart resample of y (NCHW bf16) into f32 NCHW output
// ---------------------------------------------------------------------------
__global__ void k_lp2cart(const unsigned short* __restrict__ y, float* __restrict__ out) {
  int t = blockIdx.x * blockDim.x + threadIdx.x;        // over B*H*W
  if (t >= BB * HW) return;
  int b = t >> 16;
  int m = t & (HW - 1);
  int h = m >> 8, w = m & 255;

  const float cx = (WW - 1) * 0.5f, cy = (HH - 1) * 0.5f;
  const float rmax = sqrtf(128.5f * 128.5f * 2.0f);
  const float logrm = logf(rmax + EPSF);
  float X = (float)w - cx, Y = (float)h - cy;
  float R = sqrtf(X * X + Y * Y);
  float Th = atan2f(Y, X + EPSF);
  if (Th < 0.0f) Th += 2.0f * PI_F;
  float gx = logf(R + EPSF) * ((float)WW / logrm);      // log-r axis
  float gy = Th * ((float)HH / (2.0f * PI_F));          // theta axis

  float left = floorf(gx), top = floorf(gy);
  int li = (int)left, ti = (int)top, ri = li + 1, bi = ti + 1;
  float dl = gx - left, dr = left + 1.0f - gx;
  float dt = gy - top,  db = top + 1.0f - gy;
  bool vtl = (li >= 0) & (li < WW) & (ti >= 0) & (ti < HH);
  bool vtr = (ri >= 0) & (ri < WW) & (ti >= 0) & (ti < HH);
  bool vbl = (li >= 0) & (li < WW) & (bi >= 0) & (bi < HH);
  bool vbr = (ri >= 0) & (ri < WW) & (bi >= 0) & (bi < HH);
  int otl = vtl ? (ti * WW + li) : 0;
  int otr = vtr ? (ti * WW + ri) : 0;
  int obl = vbl ? (bi * WW + li) : 0;
  int obr = vbr ? (bi * WW + ri) : 0;
  float ftl = vtl ? db * dr : 0.0f, ftr = vtr ? db * dl : 0.0f;
  float fbl = vbl ? dt * dr : 0.0f, fbr = vbr ? dt * dl : 0.0f;

  int pb = b * CC * HW;
  #pragma unroll 4
  for (int c = 0; c < CC; ++c) {
    int p = pb + c * HW;
    float v = ftl * bf2f(y[p + otl]) + ftr * bf2f(y[p + otr])
            + fbl * bf2f(y[p + obl]) + fbr * bf2f(y[p + obr]);
    out[p + m] = v;
  }
}

// ---------------------------------------------------------------------------
extern "C" void kernel_launch(void* const* d_in, const int* in_sizes, int n_in,
                              void* d_out, int out_size, void* d_ws, size_t ws_size,
                              hipStream_t stream) {
  (void)in_sizes; (void)n_in; (void)out_size; (void)ws_size;
  const float* x      = (const float*)d_in[0];   // [16,64,256,256]
  const float* conv_w = (const float*)d_in[1];   // [64,64,3,3]
  const float* conv_b = (const float*)d_in[2];   // [64]
  float* out = (float*)d_out;

  // workspace layout (bytes)
  char* ws = (char*)d_ws;
  unsigned short* wt = (unsigned short*)ws;                         //   73,728 B
  unsigned short* lp = (unsigned short*)(ws + 73728);               // 136,323,072 B
  unsigned short* yb = (unsigned short*)(ws + 73728 + 136323072);   // 134,217,728 B

  {   // weight transform
    int n = CC * KTOT;
    k_prep_w<<<(n + 255) / 256, 256, 0, stream>>>(conv_w, wt);
  }
  {   // cart -> logpolar into padded NHWC bf16
    int n = BB * HWP * HWP;
    k_cart2lp<<<(n + 255) / 256, 256, 0, stream>>>(x, lp);
  }
  {   // WMMA conv: 65536 waves, 8 waves/block
    k_conv_wmma<<<8192, 256, 0, stream>>>(lp, wt, conv_b, yb);
  }
  {   // logpolar -> cart
    int n = BB * HW;
    k_lp2cart<<<n / 256, 256, 0, stream>>>(yb, out);
  }
}
